// SingleHeadAttention_75505525063829
// MI455X (gfx1250) — matrix operand order repaired and checked by reading, not compiled, not run on hardware
//
#include <hip/hip_runtime.h>

typedef __attribute__((ext_vector_type(16))) _Float16 v16h;
typedef __attribute__((ext_vector_type(8)))  _Float16 h8;
typedef __attribute__((ext_vector_type(8)))  float    v8f;

#define BATCH 4
#define SEQ   2048
#define DMODEL 1024

// ---------------------------------------------------------------------------
// CDNA5 async global->LDS copy (16B per lane), tracked by ASYNCcnt.
// ---------------------------------------------------------------------------
__device__ __forceinline__ void async_copy16(unsigned ldsOff, const void* g) {
  asm volatile("global_load_async_to_lds_b128 %0, %1, off"
               :: "v"(ldsOff), "v"(g) : "memory");
}
__device__ __forceinline__ void wait_async_le4() {
  asm volatile("s_wait_asynccnt 0x4" ::: "memory");
}
__device__ __forceinline__ void wait_async_le0() {
  asm volatile("s_wait_asynccnt 0x0" ::: "memory");
}

// ---------------------------------------------------------------------------
// f32 -> f16 conversion
// ---------------------------------------------------------------------------
__global__ void cvt_f32_f16(const float* __restrict__ src,
                            _Float16* __restrict__ dst, int n) {
  int i = blockIdx.x * blockDim.x + threadIdx.x;
  int stride = gridDim.x * blockDim.x;
  for (; i < n; i += stride) dst[i] = (_Float16)src[i];
}

// ---------------------------------------------------------------------------
// f32 [D,D] -> f16 transposed [D,D]: dst[n*D + k] = src[k*D + n]
// 32x32 LDS tile, 32x8 threads.
// ---------------------------------------------------------------------------
__global__ void __launch_bounds__(256)
cvt_transpose_f32_f16(const float* __restrict__ src, _Float16* __restrict__ dst) {
  __shared__ float t[32][33];
  const int bx = blockIdx.x * 32, by = blockIdx.y * 32;
  const int tx = threadIdx.x, ty = threadIdx.y;
#pragma unroll
  for (int j = 0; j < 32; j += 8)
    t[ty + j][tx] = src[(size_t)(by + ty + j) * DMODEL + (bx + tx)];
  __syncthreads();
#pragma unroll
  for (int j = 0; j < 32; j += 8)
    dst[(size_t)(bx + ty + j) * DMODEL + (by + tx)] = (_Float16)t[tx][ty + j];
}

// ---------------------------------------------------------------------------
// Tiled WMMA GEMM:  C[M,N] = A[M,K] * B^T  where B is given as [N,K] row-major.
// Block tile 128x128, K-step 32, 256 threads = 8 waves, wave tile 32x64.
// LDS double-buffered; tiles staged with global_load_async_to_lds_b128.
//   OUT_F32  : write f32 (else f16)
//   OUT_TRANS: write C transposed per batch as [z][N][SEQ] (f16, vector stores)
// ---------------------------------------------------------------------------
template<bool OUT_F32, bool OUT_TRANS>
__global__ void __launch_bounds__(256)
gemm_wmma(const _Float16* __restrict__ A, const _Float16* __restrict__ Bm,
          void* __restrict__ Cv, int M, int N, int K,
          long long strideA, long long strideB, long long strideC) {
  constexpr int LDSS  = 40;            // 32 + 8 pad halves; 80B row (16B-aligned rows)
  constexpr int TILEH = 128 * LDSS;    // halves per staged matrix (5120)
  __shared__ _Float16 sm[2][2 * TILEH];

  const int tid  = threadIdx.x;
  const int lane = tid & 31;
  const int w    = tid >> 5;
  const int wm   = (w & 3) * 32;
  const int wn   = (w >> 2) * 64;
  const int bm0  = blockIdx.y * 128;
  const int bn0  = blockIdx.x * 128;
  const int z    = blockIdx.z;

  const _Float16* Ab = A  + (size_t)z * (size_t)strideA;
  const _Float16* Bb = Bm + (size_t)z * (size_t)strideB;

  // per-thread async staging addresses: 512 chunks of 16B per matrix,
  // thread handles rows (tid>>2) and 64+(tid>>2), 8-half chunk (tid&3)*8
  const int crow = tid >> 2;
  const int cch  = (tid & 3) << 3;
  const _Float16* aSrc = Ab + (size_t)(bm0 + crow) * K + cch;
  const _Float16* bSrc = Bb + (size_t)(bn0 + crow) * K + cch;
  const unsigned smBase   = (unsigned)(unsigned long long)&sm[0][0];
  const unsigned ldsOffA  = (unsigned)((crow * LDSS + cch) * 2);
  constexpr unsigned ROWS64 = 64u * LDSS * 2u;   // byte offset of 2nd chunk
  constexpr unsigned BOFF   = (unsigned)TILEH * 2u;
  constexpr unsigned BUFSZ  = 4u * (unsigned)TILEH;

  const v8f vzero = {0.f,0.f,0.f,0.f,0.f,0.f,0.f,0.f};
  v8f acc[2][4];
#pragma unroll
  for (int i = 0; i < 2; ++i)
#pragma unroll
    for (int j = 0; j < 4; ++j) acc[i][j] = vzero;

  const int l15 = lane & 15;
  const int kh  = (lane >> 4) << 3;    // A frag half-select: 0 or 8
  const int kb  = (lane >> 4) << 4;    // 0 or 16; unused (B frag also uses kh)

  auto stage = [&](int buf, int k0) {
    const unsigned base = smBase + (unsigned)buf * BUFSZ;
    async_copy16(base + ldsOffA,                aSrc + k0);
    async_copy16(base + ldsOffA + ROWS64,       aSrc + (size_t)64 * K + k0);
    async_copy16(base + BOFF + ldsOffA,         bSrc + k0);
    async_copy16(base + BOFF + ldsOffA + ROWS64, bSrc + (size_t)64 * K + k0);
  };

  const int nk = K >> 5;
  stage(0, 0);
  int cur = 0;
  for (int kt = 0; kt < nk; ++kt) {
    if (kt + 1 < nk) { stage(cur ^ 1, (kt + 1) << 5); wait_async_le4(); }
    else             { wait_async_le0(); }
    __syncthreads();

    const _Float16* sA = &sm[cur][0];
    const _Float16* sB = &sm[cur][TILEH];

    v16h av[2];
#pragma unroll
    for (int mt = 0; mt < 2; ++mt) {
      int r = wm + mt * 16 + l15;
      h8 lo = *(const h8*)&sA[r * LDSS + kh];
      h8 hi = *(const h8*)&sA[r * LDSS + 16 + kh];
      av[mt] = __builtin_shufflevector(lo, hi, 0,1,2,3,4,5,6,7,8,9,10,11,12,13,14,15);
    }
    v16h bv[4];
#pragma unroll
    for (int nt = 0; nt < 4; ++nt) {
      int c = wn + nt * 16 + l15;
      h8 lo = *(const h8*)&sB[c * LDSS + kh];
      h8 hi = *(const h8*)&sB[c * LDSS + 16 + kh];
      bv[nt] = __builtin_shufflevector(lo, hi, 0,1,2,3,4,5,6,7,8,9,10,11,12,13,14,15);
    }

#pragma unroll
    for (int mt = 0; mt < 2; ++mt)
#pragma unroll
      for (int nt = 0; nt < 4; ++nt)
        acc[mt][nt] = __builtin_amdgcn_wmma_f32_16x16x32_f16(
            false, av[mt], false, bv[nt], (short)0, acc[mt][nt], false, false);

    __syncthreads();
    cur ^= 1;
  }

  // ---- epilogue (C layout: VGPR r -> row (lane>>4)*8 + r, col lane&15) ----
  const int rb = (lane >> 4) << 3;
  if constexpr (OUT_TRANS) {
    // C stored as [z][N][SEQ] with z derived from the folded-M row index.
    _Float16* C = (_Float16*)Cv;
#pragma unroll
    for (int mt = 0; mt < 2; ++mt) {
      const int m0 = bm0 + wm + mt * 16 + rb;   // multiple of 8, one batch per block
      const int zt = m0 >> 11;                  // m0 / SEQ
      const int st = m0 & (SEQ - 1);
#pragma unroll
      for (int nt = 0; nt < 4; ++nt) {
        const int n = bn0 + wn + nt * 16 + l15;
        h8 o;
#pragma unroll
        for (int r = 0; r < 8; ++r) o[r] = (_Float16)acc[mt][nt][r];
        *(h8*)&C[((size_t)zt * DMODEL + n) * SEQ + st] = o;
      }
    }
  } else if constexpr (OUT_F32) {
    float* C = (float*)Cv + (size_t)z * (size_t)strideC;
#pragma unroll
    for (int mt = 0; mt < 2; ++mt)
#pragma unroll
      for (int nt = 0; nt < 4; ++nt)
#pragma unroll
        for (int r = 0; r < 8; ++r)
          C[(size_t)(bm0 + wm + mt * 16 + rb + r) * N + (bn0 + wn + nt * 16 + l15)] =
              acc[mt][nt][r];
  } else {
    _Float16* C = (_Float16*)Cv + (size_t)z * (size_t)strideC;
#pragma unroll
    for (int mt = 0; mt < 2; ++mt)
#pragma unroll
      for (int nt = 0; nt < 4; ++nt)
#pragma unroll
        for (int r = 0; r < 8; ++r)
          C[(size_t)(bm0 + wm + mt * 16 + rb + r) * N + (bn0 + wn + nt * 16 + l15)] =
              (_Float16)acc[mt][nt][r];
  }
}

// ---------------------------------------------------------------------------
// scale + mask + softmax over a row of SEQ f16 scores; writes P (f16) in place.
// ---------------------------------------------------------------------------
__global__ void __launch_bounds__(256)
softmax_mask(_Float16* __restrict__ Sc, const unsigned char* __restrict__ mask) {
  const int q = blockIdx.x, b = blockIdx.y, tid = threadIdx.x;
  _Float16* row = Sc + ((size_t)b * SEQ + q) * SEQ;
  const unsigned char* mrow = mask + ((size_t)b * SEQ + q) * SEQ;

  float v[8];
#pragma unroll
  for (int j = 0; j < 8; ++j) {
    int idx = tid + j * 256;
    v[j] = mrow[idx] ? (float)row[idx] * 0.03125f : -1e10f;  // 1/sqrt(1024)
  }

  __shared__ float red[256];
  float m = v[0];
#pragma unroll
  for (int j = 1; j < 8; ++j) m = fmaxf(m, v[j]);
  red[tid] = m;
  __syncthreads();
  for (int s = 128; s > 0; s >>= 1) {
    if (tid < s) red[tid] = fmaxf(red[tid], red[tid + s]);
    __syncthreads();
  }
  m = red[0];
  __syncthreads();

  float e[8], sum = 0.f;
#pragma unroll
  for (int j = 0; j < 8; ++j) { e[j] = __expf(v[j] - m); sum += e[j]; }
  red[tid] = sum;
  __syncthreads();
  for (int s = 128; s > 0; s >>= 1) {
    if (tid < s) red[tid] += red[tid + s];
    __syncthreads();
  }
  const float inv = 1.0f / red[0];
#pragma unroll
  for (int j = 0; j < 8; ++j)
    row[tid + j * 256] = (_Float16)(e[j] * inv);
}

// ---------------------------------------------------------------------------
extern "C" void kernel_launch(void* const* d_in, const int* in_sizes, int n_in,
                              void* d_out, int out_size, void* d_ws, size_t ws_size,
                              hipStream_t stream) {
  const float* x          = (const float*)d_in[0];
  const unsigned char* mk = (const unsigned char*)d_in[1];   // jnp.bool_ -> 1 byte
  const float* Wq         = (const float*)d_in[2];
  const float* Wk         = (const float*)d_in[3];
  const float* Wv         = (const float*)d_in[4];
  const float* Wo         = (const float*)d_in[5];
  float* out              = (float*)d_out;

  const size_t nXD = (size_t)BATCH * SEQ * DMODEL;   // 8,388,608
  const size_t nDD = (size_t)DMODEL * DMODEL;        // 1,048,576
  const size_t nSS = (size_t)BATCH * SEQ * SEQ;      // 16,777,216

  char* p = (char*)d_ws;
  _Float16* xh   = (_Float16*)p; p += nXD * 2;
  _Float16* wqT  = (_Float16*)p; p += nDD * 2;
  _Float16* wkT  = (_Float16*)p; p += nDD * 2;
  _Float16* wvT  = (_Float16*)p; p += nDD * 2;
  _Float16* woT  = (_Float16*)p; p += nDD * 2;
  _Float16* Qh   = (_Float16*)p; p += nXD * 2;
  _Float16* Kh   = (_Float16*)p; p += nXD * 2;
  _Float16* VhT  = (_Float16*)p; p += nXD * 2;   // [BATCH][DMODEL][SEQ]
  _Float16* Sc   = (_Float16*)p; p += nSS * 2;
  _Float16* Oh   = xh;   // xh is dead after the QKV projections; reuse it

  cvt_f32_f16<<<2048, 256, 0, stream>>>(x, xh, (int)nXD);
  dim3 tb(32, 8), tg(DMODEL / 32, DMODEL / 32);
  cvt_transpose_f32_f16<<<tg, tb, 0, stream>>>(Wq, wqT);
  cvt_transpose_f32_f16<<<tg, tb, 0, stream>>>(Wk, wkT);
  cvt_transpose_f32_f16<<<tg, tb, 0, stream>>>(Wv, wvT);
  cvt_transpose_f32_f16<<<tg, tb, 0, stream>>>(Wo, woT);

  dim3 blk(256);

  // Q/K projections: [8192,1024] = [8192,1024] x WqT/WkT^T
  dim3 gqkv(DMODEL / 128, (BATCH * SEQ) / 128, 1);
  gemm_wmma<false, false><<<gqkv, blk, 0, stream>>>(xh, wqT, Qh, BATCH * SEQ, DMODEL, DMODEL, 0, 0, 0);
  gemm_wmma<false, false><<<gqkv, blk, 0, stream>>>(xh, wkT, Kh, BATCH * SEQ, DMODEL, DMODEL, 0, 0, 0);
  // V projection, stored transposed per batch: VhT[z][d][s]
  gemm_wmma<false, true ><<<gqkv, blk, 0, stream>>>(xh, wvT, VhT, BATCH * SEQ, DMODEL, DMODEL, 0, 0, 0);

  // scores = Q x K^T per batch: [2048,2048]
  dim3 gs(SEQ / 128, SEQ / 128, BATCH);
  gemm_wmma<false, false><<<gs, blk, 0, stream>>>(
      Qh, Kh, Sc, SEQ, SEQ, DMODEL,
      (long long)SEQ * DMODEL, (long long)SEQ * DMODEL, (long long)SEQ * SEQ);

  // mask + scale + softmax (in place, f16 P)
  dim3 gsm(SEQ, BATCH);
  softmax_mask<<<gsm, blk, 0, stream>>>(Sc, mk);

  // attn_out = P x V per batch = P x (V^T)^T : [2048,1024]
  dim3 gpv(DMODEL / 128, SEQ / 128, BATCH);
  gemm_wmma<false, false><<<gpv, blk, 0, stream>>>(
      Sc, VhT, Oh, SEQ, DMODEL, SEQ,
      (long long)SEQ * SEQ, (long long)DMODEL * SEQ, (long long)SEQ * DMODEL);

  // out = attn_out x Wo : [8192,1024], f32 output
  dim3 go(DMODEL / 128, (BATCH * SEQ) / 128, 1);
  gemm_wmma<true, false><<<go, blk, 0, stream>>>(Oh, woT, (void*)out, BATCH * SEQ, DMODEL, DMODEL, 0, 0, 0);
}
